// PointNetPlusEncoder_4690104287347
// MI455X (gfx1250) — compile-verified, matrix operand order checked by reading, not run on hardware
//
#include <hip/hip_runtime.h>
#include <hip/hip_bf16.h>

#define B_ 8
#define N0 2048
#define GN_G 8
#define GN_EPS 1e-5f
#define ZDIM 128
#define CDIV(a,b) (((a)+(b)-1)/(b))

typedef _Float16 v16h __attribute__((ext_vector_type(16)));
typedef float    v8f  __attribute__((ext_vector_type(8)));

// ---------------------------------------------------------------- WMMA helpers
__device__ __forceinline__ v8f wmma_f16(v16h a, v16h b, v8f c) {
  // D = A(16x32 f16) * B(32x16 f16) + C(16x16 f32)
  return __builtin_amdgcn_wmma_f32_16x16x32_f16(false, a, false, b, (short)0, c, false, false);
}
// K index held by element e (e=0..15) of a 16-bit A/B fragment for a lane in
// half `half` (0: lanes 0-15, 1: lanes 16-31), per ISA 7.12.2 16-bit layout.
__device__ __forceinline__ int frag_k(int e, int half) {
  int i = e >> 1;
  int base = (i < 4) ? (2 * i) : (16 + 2 * (i - 4));
  return base + half * 8 + (e & 1);
}
__device__ __forceinline__ float swishf(float x) { return x / (1.f + __expf(-x)); }

// ---------------------------------------------------------------- packing
__global__ void pack_a_kernel(const float* __restrict__ W, _Float16* __restrict__ Ap,
                              int M, int K, int Kpad, long tot) {
  long i = (long)blockIdx.x * blockDim.x + threadIdx.x;
  if (i >= tot) return;
  int m = (int)(i / Kpad), k = (int)(i % Kpad);
  float v = (m < M && k < K) ? W[(long)m * K + k] : 0.f;
  Ap[i] = (_Float16)v;
}

// X: per-batch [K][Np] f32 -> Bp per-batch [Npad][Kpad] f16 (N-major, K contiguous)
__global__ void pack_bT_kernel(const float* __restrict__ X, _Float16* __restrict__ Bp,
                               int K, int Np, int Kpad, int Npad) {
  long per = (long)Npad * Kpad;
  long i = (long)blockIdx.x * blockDim.x + threadIdx.x;
  if (i >= per) return;
  const float* Xb = X + (long)blockIdx.z * K * Np;
  _Float16* Bb = Bp + (long)blockIdx.z * per;
  int n = (int)(i / Kpad), k = (int)(i % Kpad);
  float v = (n < Np && k < K) ? Xb[(long)k * Np + n] : 0.f;
  Bb[i] = (_Float16)v;
}

// ---------------------------------------------------------------- generic WMMA GEMM
// C[b][m][n] = sum_k Ap[m][k] * Bp[b][n][k] + bias[m]
__global__ __launch_bounds__(128)
void gemm_wmma_kernel(const _Float16* __restrict__ Ap, const _Float16* __restrict__ Bp,
                      const float* __restrict__ bias, float* __restrict__ C,
                      int M, int N, int Kpad, long strideB, long strideC) {
  const _Float16* Bb = Bp + (long)blockIdx.z * strideB;
  float* Cb = C + (long)blockIdx.z * strideC;
  int lane = threadIdx.x & 31, wave = threadIdx.x >> 5;
  int half = lane >> 4, l16 = lane & 15;
  int mtile = blockIdx.y * 32;
  int nb = blockIdx.x * 64 + wave * 16;
  v8f acc0 = {}; v8f acc1 = {};
  const _Float16* arow0 = Ap + (long)(mtile + l16) * Kpad;
  const _Float16* arow1 = Ap + (long)(mtile + 16 + l16) * Kpad;
  const _Float16* brow  = Bb + (long)(nb + l16) * Kpad;
  for (int kc = 0; kc < Kpad; kc += 32) {
    __builtin_prefetch(brow + kc + 32, 0, 0);
    v16h af0, af1, bf;
#pragma unroll
    for (int e = 0; e < 16; e += 2) {
      int k = kc + frag_k(e, half);
      af0[e] = arow0[k]; af0[e + 1] = arow0[k + 1];
      af1[e] = arow1[k]; af1[e + 1] = arow1[k + 1];
      bf[e]  = brow[k];  bf[e + 1]  = brow[k + 1];
    }
    acc0 = wmma_f16(af0, bf, acc0);
    acc1 = wmma_f16(af1, bf, acc1);
  }
  int n = nb + l16;
  if (n < N) {
#pragma unroll
    for (int v = 0; v < 8; ++v) {
      int m0 = mtile + v + 8 * half, m1 = m0 + 16;
      if (m0 < M) Cb[(long)m0 * N + n] = acc0[v] + bias[m0];
      if (m1 < M) Cb[(long)m1 * N + n] = acc1[v] + bias[m1];
    }
  }
}

// ---------------------------------------------------------------- implicit-GEMM conv3d (3x3x3 SAME)
__global__ __launch_bounds__(128)
void conv3d_wmma_kernel(const float* __restrict__ in, const _Float16* __restrict__ Ap,
                        const float* __restrict__ bias, float* __restrict__ out,
                        int Cin, int Cout, int r, int Kdim, int Kpad) {
  int b = blockIdx.y;
  int r3 = r * r * r;
  const float* inb = in + (long)b * Cin * r3;
  float* outb = out + (long)b * Cout * r3;
  int ntile = blockIdx.x * 64;
  int tid = threadIdx.x;
  int lane = tid & 31, wave = tid >> 5;
  int half = lane >> 4, l16 = lane & 15;
  __shared__ _Float16 Bt[64][34];   // 34-half row stride (17 banks) avoids conflicts
  v8f acc0 = {}; v8f acc1 = {};
  for (int kc = 0; kc < Kpad; kc += 32) {
    __syncthreads();
    // stage im2col tile: 64 voxels x 32 K into LDS, 16 elements per thread
#pragma unroll 4
    for (int j = 0; j < 16; ++j) {
      int idx = tid * 16 + j;
      int nl = idx >> 5, kl = idx & 31;
      int k = kc + kl;
      int nvox = ntile + nl;
      float val = 0.f;
      if (k < Kdim && nvox < r3) {
        int ci = k / 27, rem = k - ci * 27;
        int dz = rem / 9 - 1, dy = (rem / 3) % 3 - 1, dx = rem % 3 - 1;
        int z = nvox / (r * r), y = (nvox / r) % r, x = nvox % r;
        int zz = z + dz, yy = y + dy, xx = x + dx;
        if ((unsigned)zz < (unsigned)r && (unsigned)yy < (unsigned)r && (unsigned)xx < (unsigned)r)
          val = inb[(long)ci * r3 + ((long)zz * r + yy) * r + xx];
      }
      Bt[nl][kl] = (_Float16)val;
    }
    __syncthreads();
    const _Float16* arow0 = Ap + (long)l16 * Kpad + kc;
    const _Float16* arow1 = Ap + (long)(16 + l16) * Kpad + kc;
    const _Float16* brow  = &Bt[wave * 16 + l16][0];
    v16h af0, af1, bf;
#pragma unroll
    for (int e = 0; e < 16; e += 2) {
      int k = frag_k(e, half);
      af0[e] = arow0[k]; af0[e + 1] = arow0[k + 1];
      af1[e] = arow1[k]; af1[e + 1] = arow1[k + 1];
      bf[e]  = brow[k];  bf[e + 1]  = brow[k + 1];
    }
    acc0 = wmma_f16(af0, bf, acc0);
    acc1 = wmma_f16(af1, bf, acc1);
  }
  int n = ntile + wave * 16 + l16;
  if (n < r3) {
#pragma unroll
    for (int v = 0; v < 8; ++v) {
      int m0 = v + 8 * half, m1 = m0 + 16;
      if (m0 < Cout) outb[(long)m0 * r3 + n] = acc0[v] + bias[m0];
      if (m1 < Cout) outb[(long)m1 * r3 + n] = acc1[v] + bias[m1];
    }
  }
}

// ---------------------------------------------------------------- flash attention (c=32)
// q,k,v,o: per-batch [32][S] f32.  softmax(q^T k) then o = v * w^T, no scale.
__global__ __launch_bounds__(32)
void attn_wmma_kernel(const float* __restrict__ q, const float* __restrict__ k,
                      const float* __restrict__ v, float* __restrict__ o, int S) {
  int b = blockIdx.y;
  const float* qb = q + (long)b * 32 * S;
  const float* kb = k + (long)b * 32 * S;
  const float* vb = v + (long)b * 32 * S;
  float* ob = o + (long)b * 32 * S;
  int qbase = blockIdx.x * 16;
  int lane = threadIdx.x;
  int half = lane >> 4, l16 = lane & 15;
  __shared__ _Float16 Plds[16][32];
  // Q fragment: A-layout, M = query = l16, K = channel
  v16h qa;
#pragma unroll
  for (int e = 0; e < 16; e += 2) {
    int c = frag_k(e, half);
    qa[e]     = (_Float16)qb[(long)c * S + qbase + l16];
    qa[e + 1] = (_Float16)qb[(long)(c + 1) * S + qbase + l16];
  }
  float mx[8], li[8];
#pragma unroll
  for (int vv = 0; vv < 8; ++vv) { mx[vv] = -3e38f; li[vv] = 0.f; }
  v8f acc0 = {}; v8f acc1 = {};
  for (int m0 = 0; m0 < S; m0 += 32) {
    // K fragments: B-layout, K = channel, N = key
    v16h kf0, kf1;
#pragma unroll
    for (int e = 0; e < 16; e += 2) {
      int c = frag_k(e, half);
      int mA = m0 + l16, mB = m0 + 16 + l16;
      kf0[e]     = (_Float16)kb[(long)c * S + mA];
      kf0[e + 1] = (_Float16)kb[(long)(c + 1) * S + mA];
      kf1[e]     = (_Float16)kb[(long)c * S + mB];
      kf1[e + 1] = (_Float16)kb[(long)(c + 1) * S + mB];
    }
    v8f z = {};
    v8f s0 = wmma_f16(qa, kf0, z);
    v8f s1 = wmma_f16(qa, kf1, z);
    // online softmax: rows n = vv + 8*half, cols = keys (l16)
#pragma unroll
    for (int vv = 0; vv < 8; ++vv) {
      float cm = fmaxf(s0[vv], s1[vv]);
      cm = fmaxf(cm, __shfl_xor(cm, 1));
      cm = fmaxf(cm, __shfl_xor(cm, 2));
      cm = fmaxf(cm, __shfl_xor(cm, 4));
      cm = fmaxf(cm, __shfl_xor(cm, 8));
      float nm = fmaxf(mx[vv], cm);
      float sc = __expf(mx[vv] - nm);
      float p0 = __expf(s0[vv] - nm), p1 = __expf(s1[vv] - nm);
      float cs = p0 + p1;
      cs += __shfl_xor(cs, 1);
      cs += __shfl_xor(cs, 2);
      cs += __shfl_xor(cs, 4);
      cs += __shfl_xor(cs, 8);
      li[vv] = li[vv] * sc + cs;
      mx[vv] = nm;
      acc0[vv] *= sc; acc1[vv] *= sc;
      Plds[vv + 8 * half][l16] = (_Float16)p0;
      Plds[vv + 8 * half][16 + l16] = (_Float16)p1;
    }
    __syncthreads();
    // P as A (16 queries x 32 keys), V^T as B (32 keys x 16 channels)
    v16h pf, vf0, vf1;
#pragma unroll
    for (int e = 0; e < 16; e += 2) {
      int kk = frag_k(e, half);
      pf[e] = Plds[l16][kk]; pf[e + 1] = Plds[l16][kk + 1];
      vf0[e]     = (_Float16)vb[(long)l16 * S + m0 + kk];
      vf0[e + 1] = (_Float16)vb[(long)l16 * S + m0 + kk + 1];
      vf1[e]     = (_Float16)vb[(long)(16 + l16) * S + m0 + kk];
      vf1[e + 1] = (_Float16)vb[(long)(16 + l16) * S + m0 + kk + 1];
    }
    acc0 = wmma_f16(pf, vf0, acc0);
    acc1 = wmma_f16(pf, vf1, acc1);
    __syncthreads();
  }
#pragma unroll
  for (int vv = 0; vv < 8; ++vv) {
    int nq = qbase + vv + 8 * half;
    float inv = 1.f / li[vv];
    ob[(long)l16 * S + nq] = acc0[vv] * inv;
    ob[(long)(16 + l16) * S + nq] = acc1[vv] * inv;
  }
}

// ---------------------------------------------------------------- scalar / VALU kernels
__global__ void fill_zero_kernel(float* p, long n) {
  long t = (long)blockIdx.x * blockDim.x + threadIdx.x;
  if (t < n) p[t] = 0.f;
}
__global__ void transpose_in_kernel(const float* x, float* coords, int n) {
  int t = blockIdx.x * blockDim.x + threadIdx.x;
  if (t >= B_ * n) return;
  int b = t / n, i = t % n;
#pragma unroll
  for (int d = 0; d < 3; ++d)
    coords[((long)b * 3 + d) * n + i] = x[((long)b * n + i) * 3 + d];
}
__global__ void mean3_kernel(const float* crd, float* cmean, int n) {
  int d = blockIdx.x, b = blockIdx.y;
  const float* p = crd + ((long)b * 3 + d) * n;
  __shared__ float rs[256];
  float s = 0.f;
  for (int i = threadIdx.x; i < n; i += 256) s += p[i];
  rs[threadIdx.x] = s; __syncthreads();
  for (int st = 128; st > 0; st >>= 1) {
    if (threadIdx.x < st) rs[threadIdx.x] += rs[threadIdx.x + st];
    __syncthreads();
  }
  if (threadIdx.x == 0) cmean[b * 3 + d] = rs[0] / (float)n;
}
__global__ void maxnorm_kernel(const float* crd, const float* cmean, float* mnorm, int n) {
  int b = blockIdx.x;
  __shared__ float rs[256];
  float m0 = cmean[b * 3], m1 = cmean[b * 3 + 1], m2 = cmean[b * 3 + 2];
  float mx = 0.f;
  for (int i = threadIdx.x; i < n; i += 256) {
    float dx = crd[((long)b * 3) * n + i] - m0;
    float dy = crd[((long)b * 3 + 1) * n + i] - m1;
    float dz = crd[((long)b * 3 + 2) * n + i] - m2;
    mx = fmaxf(mx, sqrtf(dx * dx + dy * dy + dz * dz));
  }
  rs[threadIdx.x] = mx; __syncthreads();
  for (int st = 128; st > 0; st >>= 1) {
    if (threadIdx.x < st) rs[threadIdx.x] = fmaxf(rs[threadIdx.x], rs[threadIdx.x + st]);
    __syncthreads();
  }
  if (threadIdx.x == 0) mnorm[b] = rs[0];
}
__global__ void normalize_kernel(const float* crd, const float* cmean, const float* mnorm,
                                 float* normo, int n, int r) {
  int t = blockIdx.x * blockDim.x + threadIdx.x;
  if (t >= B_ * n) return;
  int b = t / n, i = t % n;
  float mx = fmaxf(mnorm[b], 1e-12f);
#pragma unroll
  for (int d = 0; d < 3; ++d) {
    float v = (crd[((long)b * 3 + d) * n + i] - cmean[b * 3 + d]) / (2.f * mx) + 0.5f;
    v = v * (float)r;
    v = fminf(fmaxf(v, 0.f), (float)(r - 1));
    normo[((long)b * 3 + d) * n + i] = v;
  }
}
__global__ void scatter_kernel(const float* feats, const float* norm, float* grid,
                               float* counts, int n, int C, int r, int r3) {
  int t = blockIdx.x * blockDim.x + threadIdx.x;
  if (t >= B_ * n) return;
  int b = t / n, i = t % n;
  int v0 = (int)roundf(norm[((long)b * 3) * n + i]);
  int v1 = (int)roundf(norm[((long)b * 3 + 1) * n + i]);
  int v2 = (int)roundf(norm[((long)b * 3 + 2) * n + i]);
  int idx = (v0 * r + v1) * r + v2;
  atomicAdd(&counts[(long)b * r3 + idx], 1.f);
  for (int c = 0; c < C; ++c)
    atomicAdd(&grid[((long)b * C + c) * r3 + idx], feats[((long)b * C + c) * n + i]);
}
__global__ void divcnt_kernel(float* grid, const float* counts, int C, long r3, long total) {
  long t = (long)blockIdx.x * blockDim.x + threadIdx.x;
  if (t >= total) return;
  long per = (long)C * r3;
  long b = t / per;
  long idx = (t % per) % r3;
  grid[t] /= fmaxf(counts[b * r3 + idx], 1.f);
}
__global__ void groupnorm_kernel(const float* x, float* y, const float* gamma,
                                 const float* beta, int C, long S, int dosw) {
  int g = blockIdx.x, b = blockIdx.y;
  int cpg = C / GN_G;
  const float* xb = x + ((long)b * C + (long)g * cpg) * S;
  float* yb = y + ((long)b * C + (long)g * cpg) * S;
  long tot = (long)cpg * S;
  float s = 0.f, s2 = 0.f;
  for (long i = threadIdx.x; i < tot; i += 256) { float v = xb[i]; s += v; s2 += v * v; }
  __shared__ float rs[256], rq[256];
  rs[threadIdx.x] = s; rq[threadIdx.x] = s2; __syncthreads();
  for (int st = 128; st > 0; st >>= 1) {
    if (threadIdx.x < st) { rs[threadIdx.x] += rs[threadIdx.x + st]; rq[threadIdx.x] += rq[threadIdx.x + st]; }
    __syncthreads();
  }
  float mu = rs[0] / (float)tot;
  float var = rq[0] / (float)tot - mu * mu;
  float inv = rsqrtf(var + GN_EPS);
  for (long i = threadIdx.x; i < tot; i += 256) {
    int cl = (int)(i / S);
    float val = (xb[i] - mu) * inv * gamma[g * cpg + cl] + beta[g * cpg + cl];
    yb[i] = dosw ? swishf(val) : val;
  }
}
__global__ void se_reduce_kernel(const float* h, float* mean, long S) {
  int c = blockIdx.x, b = blockIdx.y, C = gridDim.x;
  const float* p = h + ((long)b * C + c) * S;
  __shared__ float rs[256];
  float s = 0.f;
  for (long i = threadIdx.x; i < S; i += 256) s += p[i];
  rs[threadIdx.x] = s; __syncthreads();
  for (int st = 128; st > 0; st >>= 1) {
    if (threadIdx.x < st) rs[threadIdx.x] += rs[threadIdx.x + st];
    __syncthreads();
  }
  if (threadIdx.x == 0) mean[b * C + c] = rs[0] / (float)S;
}
__global__ void se_mlp_kernel(const float* mean, const float* se1, const float* se2,
                              float* s, int C, int H) {
  int b = blockIdx.x;
  __shared__ float hid[16];
  const float* mb = mean + b * C;
  if ((int)threadIdx.x < H) {
    float a = 0.f;
    for (int c = 0; c < C; ++c) a += mb[c] * se1[threadIdx.x * C + c];
    hid[threadIdx.x] = fmaxf(a, 0.f);
  }
  __syncthreads();
  if ((int)threadIdx.x < C) {
    float a = 0.f;
    for (int h = 0; h < H; ++h) a += hid[h] * se2[threadIdx.x * H + h];
    s[b * C + threadIdx.x] = 1.f / (1.f + __expf(-a));
  }
}
__global__ void se_scale_kernel(float* h, const float* s, int C, long S, long total) {
  long t = (long)blockIdx.x * blockDim.x + threadIdx.x;
  if (t >= total) return;
  long per = (long)C * S;
  int b = (int)(t / per), c = (int)((t % per) / S);
  h[t] *= s[b * C + c];
}
__global__ void resid_swish_kernel(float* h, const float* h2, long n) {
  long t = (long)blockIdx.x * blockDim.x + threadIdx.x;
  if (t < n) h[t] = swishf(h[t] + h2[t]);
}
__global__ void add_kernel(const float* a, const float* b, float* y, long n) {
  long t = (long)blockIdx.x * blockDim.x + threadIdx.x;
  if (t < n) y[t] = a[t] + b[t];
}
__global__ void trilinear_kernel(const float* grid, const float* norm, float* out,
                                 int n, int C, int r) {
  int t = blockIdx.x * blockDim.x + threadIdx.x;
  if (t >= B_ * n) return;
  int b = t / n, i = t % n;
  long r3 = (long)r * r * r;
  float nx = norm[((long)b * 3) * n + i];
  float ny = norm[((long)b * 3 + 1) * n + i];
  float nz = norm[((long)b * 3 + 2) * n + i];
  float lx = floorf(nx), ly = floorf(ny), lz = floorf(nz);
  float fx = nx - lx, fy = ny - ly, fz = nz - lz;
  int x0 = (int)lx, y0 = (int)ly, z0 = (int)lz;
  int x1 = min(x0 + 1, r - 1), y1 = min(y0 + 1, r - 1), z1 = min(z0 + 1, r - 1);
  long i000 = ((long)x0 * r + y0) * r + z0, i001 = ((long)x0 * r + y0) * r + z1;
  long i010 = ((long)x0 * r + y1) * r + z0, i011 = ((long)x0 * r + y1) * r + z1;
  long i100 = ((long)x1 * r + y0) * r + z0, i101 = ((long)x1 * r + y0) * r + z1;
  long i110 = ((long)x1 * r + y1) * r + z0, i111 = ((long)x1 * r + y1) * r + z1;
  float w000 = (1 - fx) * (1 - fy) * (1 - fz), w001 = (1 - fx) * (1 - fy) * fz;
  float w010 = (1 - fx) * fy * (1 - fz),       w011 = (1 - fx) * fy * fz;
  float w100 = fx * (1 - fy) * (1 - fz),       w101 = fx * (1 - fy) * fz;
  float w110 = fx * fy * (1 - fz),             w111 = fx * fy * fz;
  for (int c = 0; c < C; ++c) {
    const float* g = grid + ((long)b * C + c) * r3;
    float v = g[i000] * w000 + g[i001] * w001 + g[i010] * w010 + g[i011] * w011 +
              g[i100] * w100 + g[i101] * w101 + g[i110] * w110 + g[i111] * w111;
    out[((long)b * C + c) * n + i] = v;
  }
}
__global__ void fps_kernel(const float* coords, int* idx, int n, int m) {
  int b = blockIdx.x;
  const float* cb = coords + (long)b * 3 * n;
  __shared__ float mind[2048];
  __shared__ float red[256];
  __shared__ int redi[256];
  __shared__ int last;
  if (threadIdx.x == 0) last = 0;
  for (int i = threadIdx.x; i < n; i += 256) mind[i] = 1e10f;
  __syncthreads();
  for (int it = 0; it < m; ++it) {
    int lp = last;
    if (threadIdx.x == 0) idx[b * m + it] = lp;
    float lx = cb[lp], ly = cb[n + lp], lz = cb[2 * n + lp];
    float bm = -1.f; int bi = 0;
    for (int i = threadIdx.x; i < n; i += 256) {
      float dx = cb[i] - lx, dy = cb[n + i] - ly, dz = cb[2 * n + i] - lz;
      float mv = fminf(mind[i], dx * dx + dy * dy + dz * dz);
      mind[i] = mv;
      if (mv > bm) { bm = mv; bi = i; }
    }
    red[threadIdx.x] = bm; redi[threadIdx.x] = bi; __syncthreads();
    for (int s = 128; s > 0; s >>= 1) {
      if (threadIdx.x < s && red[threadIdx.x + s] > red[threadIdx.x]) {
        red[threadIdx.x] = red[threadIdx.x + s]; redi[threadIdx.x] = redi[threadIdx.x + s];
      }
      __syncthreads();
    }
    if (threadIdx.x == 0) last = redi[0];
    __syncthreads();
  }
}
__global__ void gather_centers_kernel(const float* crd, const int* idx, float* cent, int n, int m) {
  int t = blockIdx.x * blockDim.x + threadIdx.x;
  if (t >= B_ * m) return;
  int b = t / m, i = t % m;
  int j = idx[b * m + i];
#pragma unroll
  for (int d = 0; d < 3; ++d)
    cent[((long)b * 3 + d) * m + i] = crd[((long)b * 3 + d) * n + j];
}
__global__ void ballquery_kernel(const float* cent, const float* crd, int* nidx,
                                 int m, int n, float r2, int K) {
  int t = blockIdx.x * blockDim.x + threadIdx.x;
  if (t >= B_ * m) return;
  int b = t / m, c = t % m;
  const float* co = crd + (long)b * 3 * n;
  float cx = cent[((long)b * 3) * m + c];
  float cy = cent[((long)b * 3 + 1) * m + c];
  float cz = cent[((long)b * 3 + 2) * m + c];
  int* out = nidx + ((long)b * m + c) * K;
  int cnt = 0;
  for (int j = 0; j < n && cnt < K; ++j) {
    float dx = co[j] - cx, dy = co[n + j] - cy, dz = co[2 * n + j] - cz;
    if (dx * dx + dy * dy + dz * dz <= r2) out[cnt++] = j;
  }
  int f = (cnt > 0) ? out[0] : 0;
  for (; cnt < K; ++cnt) out[cnt] = f;
}
__global__ void pack_neigh_kernel(const float* crd, const float* cent, const float* feats,
                                  const int* nidx, _Float16* Bp, int m, int n, int K,
                                  int C, int Kpad) {
  long NK = (long)m * K;
  long t = (long)blockIdx.x * blockDim.x + threadIdx.x;
  if (t >= (long)B_ * NK) return;
  int b = (int)(t / NK);
  long rem = t % NK;
  int c = (int)(rem / K);
  int j = nidx[t];
  _Float16* row = Bp + ((long)b * NK + rem) * Kpad;
#pragma unroll
  for (int d = 0; d < 3; ++d)
    row[d] = (_Float16)(crd[((long)b * 3 + d) * n + j] - cent[((long)b * 3 + d) * m + c]);
  for (int ch = 0; ch < C; ++ch)
    row[3 + ch] = (_Float16)feats[((long)b * C + ch) * n + j];
  for (int kk = 3 + C; kk < Kpad; ++kk) row[kk] = (_Float16)0.f;
}
__global__ void maxk_kernel(const float* h, float* out, int C, int m, int K) {
  int t = blockIdx.x * blockDim.x + threadIdx.x;
  if (t >= B_ * C * m) return;
  int b = t / (C * m);
  int rem = t % (C * m);
  int c = rem / m, i = rem % m;
  const float* p = h + ((long)b * C + c) * ((long)m * K) + (long)i * K;
  float mx = -3e38f;
  for (int kk = 0; kk < K; ++kk) mx = fmaxf(mx, p[kk]);
  out[((long)b * C + c) * m + i] = mx;
}
__global__ void gmax_kernel(const float* f, float* g, int C, int m) {
  int t = blockIdx.x * blockDim.x + threadIdx.x;
  if (t >= B_ * C) return;
  int b = t / C, c = t % C;
  const float* p = f + ((long)b * C + c) * m;
  float mx = -3e38f;
  for (int i = 0; i < m; ++i) mx = fmaxf(mx, p[i]);
  g[b * C + c] = mx;
}
__global__ void head_kernel(const float* g, const float* W, const float* bias,
                            float* out, int C, int O) {
  int t = blockIdx.x * blockDim.x + threadIdx.x;
  if (t >= B_ * O) return;
  int b = t / O, o = t % O;
  float a = bias[o];
  for (int c = 0; c < C; ++c) a += g[b * C + c] * W[o * C + c];
  if (o < ZDIM) out[b * ZDIM + o] = a;
  else out[(long)B_ * ZDIM + b * ZDIM + (o - ZDIM)] = a;
}

// ---------------------------------------------------------------- host orchestration
// param leaf indices (jax pytree order: sorted dict keys, lists in order), x first
enum {
  IN_X = 0, IN_HEAD_B = 1, IN_HEAD_W = 2,
  S0PV0 = 3, S0PV1 = 17, S0MLP0 = 31, S0MLP1 = 35,
  S1ATTN = 39, S1PV0 = 49, S1MLP0 = 63, S1MLP1 = 67
};
// pv block: 0 c1b,1 c1w,2 c2b,3 c2w,4 g1b,5 g1g,6 g2b,7 g2g,8 ptb,9 ptbeta,10 ptg,11 ptw,12 se1,13 se2
// attn: 0 gb,1 gg,2 kb,3 kw,4 ob,5 ow,6 qb,7 qw,8 vb,9 vw
// mlp: 0 b,1 beta,2 g,3 w

extern "C" void kernel_launch(void* const* d_in, const int* in_sizes, int n_in,
                              void* d_out, int out_size, void* d_ws, size_t ws_size,
                              hipStream_t stream) {
  (void)in_sizes; (void)n_in; (void)out_size; (void)ws_size;
  const float* x = (const float*)d_in[IN_X];
  auto P = [&](int i) { return (const float*)d_in[i]; };

  size_t off = 0;
  auto alloc = [&](size_t bytes) -> void* {
    void* p = (char*)d_ws + off;
    off += (bytes + 255) & ~(size_t)255;
    return p;
  };
  float* coords   = (float*)alloc(sizeof(float) * B_ * 3 * N0);
  float* featsA   = (float*)alloc(sizeof(float) * B_ * 64 * N0);
  float* featsB   = (float*)alloc(sizeof(float) * B_ * 64 * N0);
  float* cmean    = (float*)alloc(sizeof(float) * B_ * 3);
  float* mnorm    = (float*)alloc(sizeof(float) * B_);
  float* normbuf  = (float*)alloc(sizeof(float) * B_ * 3 * N0);
  float* grid1    = (float*)alloc(sizeof(float) * (size_t)B_ * 32 * 32768);
  float* grid2    = (float*)alloc(sizeof(float) * (size_t)B_ * 32 * 32768);
  float* grid3    = (float*)alloc(sizeof(float) * (size_t)B_ * 32 * 32768);
  float* counts   = (float*)alloc(sizeof(float) * (size_t)B_ * 32768);
  _Float16* Ap    = (_Float16*)alloc(sizeof(_Float16) * 256 * 896);
  _Float16* Bp    = (_Float16*)alloc(sizeof(_Float16) * (size_t)B_ * 32768 * 64);
  float* qbuf     = (float*)alloc(sizeof(float) * (size_t)B_ * 32 * 4096);
  float* kbuf     = (float*)alloc(sizeof(float) * (size_t)B_ * 32 * 4096);
  float* vbuf     = (float*)alloc(sizeof(float) * (size_t)B_ * 32 * 4096);
  float* aobuf    = (float*)alloc(sizeof(float) * (size_t)B_ * 32 * 4096);
  float* h2buf    = (float*)alloc(sizeof(float) * (size_t)B_ * 32 * 4096);
  float* semean   = (float*)alloc(sizeof(float) * B_ * 64);
  float* svec     = (float*)alloc(sizeof(float) * B_ * 64);
  float* vfeat    = (float*)alloc(sizeof(float) * (size_t)B_ * 32 * N0);
  float* pfeat    = (float*)alloc(sizeof(float) * (size_t)B_ * 32 * N0);
  int*   idxbuf   = (int*)alloc(sizeof(int) * B_ * 1024);
  float* centers1 = (float*)alloc(sizeof(float) * B_ * 3 * 1024);
  float* centers2 = (float*)alloc(sizeof(float) * B_ * 3 * 256);
  int*   nidx     = (int*)alloc(sizeof(int) * (size_t)B_ * 1024 * 32);
  float* mlpa     = (float*)alloc(sizeof(float) * (size_t)B_ * 32 * 32768);
  float* mlpb     = (float*)alloc(sizeof(float) * (size_t)B_ * 64 * 32768);
  float* gvec     = (float*)alloc(sizeof(float) * B_ * 64);

  auto zero = [&](float* p, long n) {
    fill_zero_kernel<<<CDIV(n, 256), 256, 0, stream>>>(p, n);
  };
  auto packA = [&](const float* W, int M, int K, int Mpad, int Kpad) {
    long tot = (long)Mpad * Kpad;
    pack_a_kernel<<<CDIV(tot, 256), 256, 0, stream>>>(W, Ap, M, K, Kpad, tot);
  };
  auto packB = [&](const float* X, int K, int Np, int Kpad, int Npad) {
    long per = (long)Npad * Kpad;
    pack_bT_kernel<<<dim3((unsigned)CDIV(per, 256), 1, B_), 256, 0, stream>>>(X, Bp, K, Np, Kpad, Npad);
  };
  auto gemm = [&](const float* bias, float* C, int M, int N, int Kpad, int Mpad, int Npad) {
    dim3 g(Npad / 64, Mpad / 32, B_);
    gemm_wmma_kernel<<<g, 128, 0, stream>>>(Ap, Bp, bias, C, M, N, Kpad,
                                            (long)Npad * Kpad, (long)M * N);
  };
  auto gnorm = [&](float* xio, const float* ga, const float* be, int C, long S, int sw) {
    groupnorm_kernel<<<dim3(GN_G, B_), 256, 0, stream>>>(xio, xio, ga, be, C, S, sw);
  };

  auto pvconv = [&](int pb, int ab, const float* fin, int Cf, const float* crd,
                    int Np, int r, float* fout) {
    int r3 = r * r * r;
    const int Cout = 32;
    // ---- voxelize
    mean3_kernel<<<dim3(3, B_), 256, 0, stream>>>(crd, cmean, Np);
    maxnorm_kernel<<<B_, 256, 0, stream>>>(crd, cmean, mnorm, Np);
    normalize_kernel<<<CDIV(B_ * Np, 256), 256, 0, stream>>>(crd, cmean, mnorm, normbuf, Np, r);
    zero(grid1, (long)B_ * Cf * r3);
    zero(counts, (long)B_ * r3);
    scatter_kernel<<<CDIV(B_ * Np, 256), 256, 0, stream>>>(fin, normbuf, grid1, counts, Np, Cf, r, r3);
    divcnt_kernel<<<CDIV((long)B_ * Cf * r3, 256), 256, 0, stream>>>(grid1, counts, Cf, r3, (long)B_ * Cf * r3);
    // ---- conv1 + GN + swish
    int Kd1 = Cf * 27, Kp1 = CDIV(Kd1, 32) * 32;
    packA(P(pb + 1), Cout, Kd1, 32, Kp1);
    conv3d_wmma_kernel<<<dim3(CDIV(r3, 64), B_), 128, 0, stream>>>(grid1, Ap, P(pb + 0), grid2, Cf, Cout, r, Kd1, Kp1);
    gnorm(grid2, P(pb + 5), P(pb + 4), Cout, r3, 1);
    // ---- conv2 + GN (+swish if no attn)
    int Kd2 = Cout * 27, Kp2 = CDIV(Kd2, 32) * 32;
    packA(P(pb + 3), Cout, Kd2, 32, Kp2);
    conv3d_wmma_kernel<<<dim3(CDIV(r3, 64), B_), 128, 0, stream>>>(grid2, Ap, P(pb + 2), grid3, Cout, Cout, r, Kd2, Kp2);
    gnorm(grid3, P(pb + 7), P(pb + 6), Cout, r3, (ab < 0) ? 1 : 0);
    // ---- attention (stage 1 only)
    if (ab >= 0) {
      int S = r3;  // 4096
      packB(grid3, 32, S, 32, S);
      packA(P(ab + 7), 32, 32, 32, 32); gemm(P(ab + 6), qbuf, 32, S, 32, 32, S);
      packA(P(ab + 3), 32, 32, 32, 32); gemm(P(ab + 2), kbuf, 32, S, 32, 32, S);
      packA(P(ab + 9), 32, 32, 32, 32); gemm(P(ab + 8), vbuf, 32, S, 32, 32, S);
      attn_wmma_kernel<<<dim3(S / 16, B_), 32, 0, stream>>>(qbuf, kbuf, vbuf, aobuf, S);
      packB(aobuf, 32, S, 32, S);
      packA(P(ab + 5), 32, 32, 32, 32); gemm(P(ab + 4), h2buf, 32, S, 32, 32, S);
      gnorm(h2buf, P(ab + 1), P(ab + 0), 32, S, 0);
      resid_swish_kernel<<<CDIV((long)B_ * 32 * S, 256), 256, 0, stream>>>(grid3, h2buf, (long)B_ * 32 * S);
    }
    // ---- squeeze-excite
    se_reduce_kernel<<<dim3(Cout, B_), 256, 0, stream>>>(grid3, semean, r3);
    se_mlp_kernel<<<B_, 64, 0, stream>>>(semean, P(pb + 12), P(pb + 13), svec, Cout, Cout / 8);
    se_scale_kernel<<<CDIV((long)B_ * Cout * r3, 256), 256, 0, stream>>>(grid3, svec, Cout, r3, (long)B_ * Cout * r3);
    // ---- devoxelize
    trilinear_kernel<<<CDIV(B_ * Np, 256), 256, 0, stream>>>(grid3, normbuf, vfeat, Np, Cout, r);
    // ---- point branch: conv1x1 + GN + swish, then add
    int KpP = CDIV(Cf, 32) * 32, NpP = CDIV(Np, 64) * 64;
    packB(fin, Cf, Np, KpP, NpP);
    packA(P(pb + 11), Cout, Cf, 32, KpP);
    gemm(P(pb + 8), pfeat, Cout, Np, KpP, 32, NpP);
    gnorm(pfeat, P(pb + 10), P(pb + 9), Cout, Np, 1);
    add_kernel<<<CDIV((long)B_ * Cout * Np, 256), 256, 0, stream>>>(vfeat, pfeat, fout, (long)B_ * Cout * Np);
  };

  auto samod = [&](int mb0, int mb1, const float* fin, int C, const float* crd, int Np,
                   int m, float rad, int K, int C2, float* fout, float* cent) {
    fps_kernel<<<B_, 256, 0, stream>>>(crd, idxbuf, Np, m);
    gather_centers_kernel<<<CDIV(B_ * m, 256), 256, 0, stream>>>(crd, idxbuf, cent, Np, m);
    ballquery_kernel<<<CDIV(B_ * m, 256), 256, 0, stream>>>(cent, crd, nidx, m, Np, rad * rad, K);
    long NK = (long)m * K;
    pack_neigh_kernel<<<CDIV((long)B_ * NK, 256), 256, 0, stream>>>(crd, cent, fin, nidx, Bp, m, Np, K, C, 64);
    packA(P(mb0 + 3), 32, 3 + C, 32, 64);
    gemm(P(mb0 + 0), mlpa, 32, (int)NK, 64, 32, (int)NK);
    gnorm(mlpa, P(mb0 + 2), P(mb0 + 1), 32, NK, 1);
    packB(mlpa, 32, (int)NK, 32, (int)NK);
    int Mp2 = CDIV(C2, 32) * 32;
    packA(P(mb1 + 3), C2, 32, Mp2, 32);
    gemm(P(mb1 + 0), mlpb, C2, (int)NK, 32, Mp2, (int)NK);
    gnorm(mlpb, P(mb1 + 2), P(mb1 + 1), C2, NK, 1);
    maxk_kernel<<<CDIV(B_ * C2 * m, 256), 256, 0, stream>>>(mlpb, fout, C2, m, K);
  };

  // ---------------- forward graph
  transpose_in_kernel<<<CDIV(B_ * N0, 256), 256, 0, stream>>>(x, coords, N0);
  // stage 0
  pvconv(S0PV0, -1, coords, 3, coords, N0, 32, featsA);
  pvconv(S0PV1, -1, featsA, 32, coords, N0, 32, featsB);
  samod(S0MLP0, S0MLP1, featsB, 32, coords, N0, 1024, 0.1f, 32, 32, featsA, centers1);
  // stage 1
  pvconv(S1PV0, S1ATTN, featsA, 32, centers1, 1024, 16, featsB);
  samod(S1MLP0, S1MLP1, featsB, 32, centers1, 1024, 256, 0.2f, 32, 64, featsA, centers2);
  // head
  gmax_kernel<<<CDIV(B_ * 64, 64), 64, 0, stream>>>(featsA, gvec, 64, 256);
  head_kernel<<<CDIV(B_ * 256, 256), 256, 0, stream>>>(gvec, P(IN_HEAD_W), P(IN_HEAD_B),
                                                       (float*)d_out, 64, 256);
}
